// RGBD_MultiHeadAttention_11647951306971
// MI455X (gfx1250) — matrix operand order repaired and checked by reading, not compiled, not run on hardware
//
#include <hip/hip_runtime.h>
#include <math.h>

#define S_  1024
#define B_  4
#define E_  256
#define H_  8
#define DH  32

typedef __attribute__((ext_vector_type(16))) __bf16 v16bf;
typedef __attribute__((ext_vector_type(8)))  float  v8f;

// ---------------------------------------------------------------------------
// WMMA fragment helpers (contiguous k order used here for both A and B, 16-bit A 16x32 / B 32x16)
// A: lanes 0-15 hold row M=lane with K {k0..k0+15};
//    lanes 16-31 hold row M=lane-16 with K {k0+16..k0+31}.
// B: lanes 0-15 hold col N=lane with K {k0..k0+15};
//    lanes 16-31 hold col N=lane-16 with K {k0+16..k0+31}  (contiguous load).
// ---------------------------------------------------------------------------
__device__ inline v16bf frag_a_from_f32(const float* arow, int lane) {
  const int koff = (lane < 16) ? 0 : 16;
  v16bf a;
#pragma unroll
  for (int i = 0; i < 8; ++i) a[i] = (__bf16)arow[koff + i];
#pragma unroll
  for (int i = 0; i < 8; ++i) a[8 + i] = (__bf16)arow[8 + koff + i];
  return a;
}

__device__ inline v16bf frag_a_from_bf16(const __bf16* arow, int lane) {
  const int koff = (lane < 16) ? 0 : 16;
  v16bf a;
#pragma unroll
  for (int i = 0; i < 8; ++i) a[i] = arow[koff + i];
#pragma unroll
  for (int i = 0; i < 8; ++i) a[8 + i] = arow[8 + koff + i];
  return a;
}

// ---------------------------------------------------------------------------
// f32 -> bf16 bulk convert (weights)
// ---------------------------------------------------------------------------
__global__ void cvt_f32_bf16(const float* __restrict__ src,
                             __bf16* __restrict__ dst, int n) {
  int i = blockIdx.x * blockDim.x + threadIdx.x;
  int stride = gridDim.x * blockDim.x;
  for (; i < n; i += stride) dst[i] = (__bf16)src[i];
}

// ---------------------------------------------------------------------------
// Fused QKV projection: y[m,n] = sum_e x[m,e] * W[n,e] + b[n]
// x is [S,B,E] (row m = s*B+b), W is [3E,E] row-major. Column block selects
// which input tensor (q/k/v use different x!). Outputs (bf16):
//   Q, K heads: [B,H,S,DH]   (A-side / B-side of QK^T, row-major in K)
//   V heads   : [B,H,DH,S]   (TRANSPOSED so PV B-fragments are contiguous)
// ---------------------------------------------------------------------------
__global__ __launch_bounds__(128)
void qkv_kernel(const float* __restrict__ xq, const float* __restrict__ xk,
                const float* __restrict__ xv, const __bf16* __restrict__ w,
                const float* __restrict__ bias, __bf16* __restrict__ qkv_out) {
  const int lane   = threadIdx.x & 31;
  const int wid    = threadIdx.x >> 5;
  const int tile_m = blockIdx.x;
  const int tile_n = blockIdx.y * 4 + wid;
  const int n0     = tile_n * 16;
  const int which  = n0 / E_;  // 0=q, 1=k, 2=v
  const float* x   = (which == 0) ? xq : (which == 1) ? xk : xv;

  const int arow = tile_m * 16 + (lane & 15);
  const int col  = n0 + (lane & 15);

  v8f acc = {};
  for (int k0 = 0; k0 < E_; k0 += 32) {
    v16bf a  = frag_a_from_f32(x + (size_t)arow * E_ + k0, lane);
    const int kb = k0 + ((lane < 16) ? 0 : 16);
    v16bf bm = *(const v16bf*)(w + (size_t)col * E_ + kb);
    acc = __builtin_amdgcn_wmma_f32_16x16x32_bf16(false, a, false, bm,
                                                  (short)0, acc, false, false);
  }

  const int f = col % E_;
  const int h = f / DH, d = f % DH;
  const float bv = bias[col];
  __bf16* base = qkv_out + (size_t)which * ((size_t)S_ * B_ * E_);
#pragma unroll
  for (int r = 0; r < 8; ++r) {
    const int m = tile_m * 16 + r + ((lane < 16) ? 0 : 8);
    const int s = m / B_, b = m % B_;
    const __bf16 v = (__bf16)(acc[r] + bv);
    if (which == 2) {
      // V transposed: [B,H,DH,S]
      base[(((size_t)b * H_ + h) * DH + d) * S_ + s] = v;
    } else {
      base[(((size_t)b * H_ + h) * S_ + s) * DH + d] = v;
    }
  }
}

// ---------------------------------------------------------------------------
// Scores: scores[q,k] = scale * sum_d qh[q,d]*kh[k,d]  (K = DH = 32).
// Register-blocked: each wave loads its Q fragment ONCE and multiplies it
// against 4 consecutive key tiles (4 WMMAs), then stores 4 16x16 f32 tiles.
// Masked positions -> -inf. Written f32 into d_out shared_* slots.
// ---------------------------------------------------------------------------
__global__ __launch_bounds__(128)
void scores_kernel(const __bf16* __restrict__ qkv_rgb,
                   const __bf16* __restrict__ qkv_dpt,
                   const unsigned char* __restrict__ kpm,
                   float* __restrict__ out_probs) {
  const int lane   = threadIdx.x & 31;
  const int wid    = threadIdx.x >> 5;
  const int z      = blockIdx.z;      // stream*32 + bh
  const int stream = z >> 5;
  const int bh     = z & 31;
  const int b      = bh / H_;
  const __bf16* qkv = stream ? qkv_dpt : qkv_rgb;
  const __bf16* qh  = qkv + (size_t)bh * S_ * DH;
  const __bf16* kh  = qkv + (size_t)S_ * B_ * E_ + (size_t)bh * S_ * DH;

  const int tile_m = blockIdx.y * 4 + wid;  // query tile (64 total)
  const int kt0    = blockIdx.x * 4;        // first of 4 key tiles
  const int qrow   = tile_m * 16 + (lane & 15);
  const int kb     = (lane < 16) ? 0 : 16;

  v16bf a = frag_a_from_bf16(qh + (size_t)qrow * DH, lane);

  v8f acc[4];
#pragma unroll
  for (int kt = 0; kt < 4; ++kt) {
    const int kk = (kt0 + kt) * 16 + (lane & 15);
    v16bf bm = *(const v16bf*)(kh + (size_t)kk * DH + kb);
    v8f c = {};
    acc[kt] = __builtin_amdgcn_wmma_f32_16x16x32_bf16(false, a, false, bm,
                                                      (short)0, c, false, false);
  }

  const float scale = 0.17677669529663687f;  // 1/sqrt(32)
  float* orow = out_probs + (size_t)stream * ((size_t)B_ * H_ * S_ * S_) +
                (size_t)bh * S_ * S_;
#pragma unroll
  for (int kt = 0; kt < 4; ++kt) {
    const int kk = (kt0 + kt) * 16 + (lane & 15);
    const bool masked = kpm[b * S_ + kk] != 0;
#pragma unroll
    for (int r = 0; r < 8; ++r) {
      const int q = tile_m * 16 + r + ((lane < 16) ? 0 : 8);
      orow[(size_t)q * S_ + kk] = masked ? -__builtin_inff() : acc[kt][r] * scale;
    }
  }
}

// ---------------------------------------------------------------------------
// Fused softmax (per row, both streams) + alpha/beta cross-stream blend.
// One block handles the same (bh, q) row of p_rgb and p_dpt; saves a full
// 536 MB read+write pass vs separate softmax/blend kernels.
// ---------------------------------------------------------------------------
__global__ __launch_bounds__(256)
void softmax_blend_kernel(float* __restrict__ probs_rgb,
                          float* __restrict__ probs_dpt,
                          const float* __restrict__ alpha_p,
                          const float* __restrict__ beta_p) {
  __shared__ float red[256];
  const size_t row = blockIdx.x;
  float* pr = probs_rgb + row * S_;
  float* pd = probs_dpt + row * S_;
  const int t = threadIdx.x;

  float rv[4], dv[4];
#pragma unroll
  for (int j = 0; j < 4; ++j) {
    rv[j] = pr[t + j * 256];
    dv[j] = pd[t + j * 256];
  }

  // ---- softmax rgb ----
  float lm = fmaxf(fmaxf(rv[0], rv[1]), fmaxf(rv[2], rv[3]));
  red[t] = lm; __syncthreads();
  for (int s = 128; s > 0; s >>= 1) {
    if (t < s) red[t] = fmaxf(red[t], red[t + s]);
    __syncthreads();
  }
  float mx = red[0]; __syncthreads();
  float ls = 0.f;
#pragma unroll
  for (int j = 0; j < 4; ++j) { rv[j] = __expf(rv[j] - mx); ls += rv[j]; }
  red[t] = ls; __syncthreads();
  for (int s = 128; s > 0; s >>= 1) {
    if (t < s) red[t] += red[t + s];
    __syncthreads();
  }
  float inv = 1.0f / red[0]; __syncthreads();
#pragma unroll
  for (int j = 0; j < 4; ++j) rv[j] *= inv;

  // ---- softmax dpt ----
  lm = fmaxf(fmaxf(dv[0], dv[1]), fmaxf(dv[2], dv[3]));
  red[t] = lm; __syncthreads();
  for (int s = 128; s > 0; s >>= 1) {
    if (t < s) red[t] = fmaxf(red[t], red[t + s]);
    __syncthreads();
  }
  mx = red[0]; __syncthreads();
  ls = 0.f;
#pragma unroll
  for (int j = 0; j < 4; ++j) { dv[j] = __expf(dv[j] - mx); ls += dv[j]; }
  red[t] = ls; __syncthreads();
  for (int s = 128; s > 0; s >>= 1) {
    if (t < s) red[t] += red[t + s];
    __syncthreads();
  }
  inv = 1.0f / red[0];
#pragma unroll
  for (int j = 0; j < 4; ++j) dv[j] *= inv;

  // ---- blend + store ----
  const float alpha = *alpha_p, beta = *beta_p;
#pragma unroll
  for (int j = 0; j < 4; ++j) {
    const float a = rv[j], d = dv[j];
    pr[t + j * 256] = (1.0f - alpha) * a + alpha * d;
    pd[t + j * 256] = (1.0f - beta) * d + beta * a;
  }
}

// ---------------------------------------------------------------------------
// PV: out[q,d] = sum_k shared[q,k] * vhT[d,k];  K = S = 1024 (32 WMMA steps).
// V is stored transposed [B,H,DH,S], so each B-fragment is a single
// contiguous 32-byte load (vs 16 strided 2-byte loads).
// A-fragment is the f32 prob row converted to bf16 in-register.
// Output stored bf16 in [B,S,E] layout (ready for out-projection).
// ---------------------------------------------------------------------------
__global__ __launch_bounds__(128)
void pv_kernel(const float* __restrict__ probs,
               const __bf16* __restrict__ qkv_rgb,
               const __bf16* __restrict__ qkv_dpt,
               __bf16* __restrict__ attn_rgb, __bf16* __restrict__ attn_dpt) {
  const int lane   = threadIdx.x & 31;
  const int wid    = threadIdx.x >> 5;
  const int z      = blockIdx.z;
  const int stream = z >> 5;
  const int bh     = z & 31;
  const int b      = bh / H_, h = bh % H_;

  const float*  p   = probs + (size_t)stream * ((size_t)B_ * H_ * S_ * S_) +
                      (size_t)bh * S_ * S_;
  const __bf16* vhT = (stream ? qkv_dpt : qkv_rgb) +
                      2 * (size_t)S_ * B_ * E_ + (size_t)bh * S_ * DH;
  __bf16* attn = stream ? attn_dpt : attn_rgb;

  const int tile_m = blockIdx.y * 4 + wid;
  const int tile_n = blockIdx.x;  // 0..1 (DH/16)
  const int qrow   = tile_m * 16 + (lane & 15);
  const int dcol   = tile_n * 16 + (lane & 15);

  v8f acc = {};
  for (int k0 = 0; k0 < S_; k0 += 32) {
    v16bf a = frag_a_from_f32(p + (size_t)qrow * S_ + k0, lane);
    const int kb = k0 + ((lane < 16) ? 0 : 16);
    v16bf bm = *(const v16bf*)(vhT + (size_t)dcol * S_ + kb);  // contiguous
    acc = __builtin_amdgcn_wmma_f32_16x16x32_bf16(false, a, false, bm,
                                                  (short)0, acc, false, false);
  }
#pragma unroll
  for (int r = 0; r < 8; ++r) {
    const int q = tile_m * 16 + r + ((lane < 16) ? 0 : 8);
    attn[((size_t)b * S_ + q) * E_ + h * DH + dcol] = (__bf16)acc[r];
  }
}

// ---------------------------------------------------------------------------
// Output projection: y[m,n] = sum_e A[m,e]*W[n,e] + b[n];  A = attn [B*S, E].
// Written f32 to d_out in [S,B,E] order.
// ---------------------------------------------------------------------------
__global__ __launch_bounds__(128)
void outproj_kernel(const __bf16* __restrict__ attn_rgb,
                    const __bf16* __restrict__ attn_dpt,
                    const __bf16* __restrict__ w_rgb,
                    const __bf16* __restrict__ w_dpt,
                    const float* __restrict__ b_rgb,
                    const float* __restrict__ b_dpt,
                    float* __restrict__ out) {
  const int lane   = threadIdx.x & 31;
  const int wid    = threadIdx.x >> 5;
  const int stream = blockIdx.z;
  const __bf16* A  = stream ? attn_dpt : attn_rgb;
  const __bf16* W  = stream ? w_dpt : w_rgb;
  const float* bias = stream ? b_dpt : b_rgb;
  float* o = out + (size_t)stream * ((size_t)S_ * B_ * E_);

  const int tile_m = blockIdx.x;
  const int tile_n = blockIdx.y * 4 + wid;
  const int m0  = tile_m * 16 + (lane & 15);
  const int col = tile_n * 16 + (lane & 15);

  v8f acc = {};
  for (int k0 = 0; k0 < E_; k0 += 32) {
    v16bf a  = frag_a_from_bf16(A + (size_t)m0 * E_ + k0, lane);
    const int kb = k0 + ((lane < 16) ? 0 : 16);
    v16bf bm = *(const v16bf*)(W + (size_t)col * E_ + kb);
    acc = __builtin_amdgcn_wmma_f32_16x16x32_bf16(false, a, false, bm,
                                                  (short)0, acc, false, false);
  }
  const float bv = bias[col];
#pragma unroll
  for (int r = 0; r < 8; ++r) {
    const int m  = tile_m * 16 + r + ((lane < 16) ? 0 : 8);
    const int bb = m / S_, s = m % S_;
    o[((size_t)s * B_ + bb) * E_ + col] = acc[r] + bv;
  }
}

// ---------------------------------------------------------------------------
extern "C" void kernel_launch(void* const* d_in, const int* in_sizes, int n_in,
                              void* d_out, int out_size, void* d_ws,
                              size_t ws_size, hipStream_t stream) {
  const float* query  = (const float*)d_in[0];
  const float* key    = (const float*)d_in[1];
  const float* value  = (const float*)d_in[2];
  const float* qd     = (const float*)d_in[3];
  const float* kd     = (const float*)d_in[4];
  const float* vd     = (const float*)d_in[5];
  const unsigned char* kpm = (const unsigned char*)d_in[6];
  const float* rgb_in_w  = (const float*)d_in[7];
  const float* rgb_in_b  = (const float*)d_in[8];
  const float* rgb_out_w = (const float*)d_in[9];
  const float* rgb_out_b = (const float*)d_in[10];
  const float* dpt_in_w  = (const float*)d_in[11];
  const float* dpt_in_b  = (const float*)d_in[12];
  const float* dpt_out_w = (const float*)d_in[13];
  const float* dpt_out_b = (const float*)d_in[14];
  const float* alpha = (const float*)d_in[15];
  const float* beta  = (const float*)d_in[16];

  const size_t EE   = (size_t)E_ * E_;        // 65536
  const size_t SBE  = (size_t)S_ * B_ * E_;   // 1,048,576
  const size_t BHSS = (size_t)B_ * H_ * S_ * S_;

  __bf16* ws        = (__bf16*)d_ws;          // ~18 MB of bf16 scratch
  __bf16* w_rgb_in  = ws;                     // 3*EE
  __bf16* w_dpt_in  = ws + 3 * EE;            // 3*EE
  __bf16* w_rgb_out = ws + 6 * EE;            // EE
  __bf16* w_dpt_out = ws + 7 * EE;            // EE
  __bf16* qkv_rgb   = ws + 8 * EE;            // 3*SBE  (q,k:[B,H,S,DH], v:[B,H,DH,S])
  __bf16* qkv_dpt   = qkv_rgb + 3 * SBE;      // 3*SBE
  __bf16* attn_rgb  = qkv_dpt + 3 * SBE;      // SBE    ([B,S,E])
  __bf16* attn_dpt  = attn_rgb + SBE;         // SBE

  // 1) weight conversion to bf16
  cvt_f32_bf16<<<256, 256, 0, stream>>>(rgb_in_w,  w_rgb_in,  (int)(3 * EE));
  cvt_f32_bf16<<<256, 256, 0, stream>>>(dpt_in_w,  w_dpt_in,  (int)(3 * EE));
  cvt_f32_bf16<<<256, 256, 0, stream>>>(rgb_out_w, w_rgb_out, (int)EE);
  cvt_f32_bf16<<<256, 256, 0, stream>>>(dpt_out_w, w_dpt_out, (int)EE);

  const dim3 blk(128);
  // 2) QKV projections (M=4096, N=768, K=256)
  qkv_kernel<<<dim3(256, 12, 1), blk, 0, stream>>>(query, key, value,
                                                   w_rgb_in, rgb_in_b, qkv_rgb);
  qkv_kernel<<<dim3(256, 12, 1), blk, 0, stream>>>(qd, kd, vd,
                                                   w_dpt_in, dpt_in_b, qkv_dpt);

  // 3) scores into d_out shared slots (pre-softmax); 4 k-tiles per wave
  float* probs_rgb = (float*)d_out + 2 * SBE;
  float* probs_dpt = probs_rgb + BHSS;
  scores_kernel<<<dim3(16, 16, 64), blk, 0, stream>>>(qkv_rgb, qkv_dpt, kpm,
                                                      probs_rgb);

  // 4) fused softmax + cross-stream blend (in place in d_out)
  softmax_blend_kernel<<<(unsigned)(B_ * H_ * S_), 256, 0, stream>>>(
      probs_rgb, probs_dpt, alpha, beta);

  // 5) PV (K = 1024, contiguous V^T fragments)
  pv_kernel<<<dim3(2, 16, 64), blk, 0, stream>>>(probs_rgb, qkv_rgb, qkv_dpt,
                                                 attn_rgb, attn_dpt);

  // 6) output projections into d_out[0 .. 2*SBE)
  outproj_kernel<<<dim3(256, 4, 2), blk, 0, stream>>>(
      attn_rgb, attn_dpt, w_rgb_out, w_dpt_out, rgb_out_b, dpt_out_b,
      (float*)d_out);
}
